// MultiHeadAttention_41566693491056
// MI455X (gfx1250) — matrix operand order repaired and checked by reading, not compiled, not run on hardware
//
#include <hip/hip_runtime.h>
#include <hip/hip_bf16.h>
#include <stdint.h>

#define EMBED 1024
#define NHEAD 16
#define HDIM  64
#define BB    4
#define SEQ   2048
#define NTOK  (BB * SEQ)   // 8192
#define SST   2052         // padded LDS stride for score rows (floats)

typedef __attribute__((ext_vector_type(16))) __bf16        v16bf;
typedef __attribute__((ext_vector_type(2)))  __bf16        v2bf;
typedef __attribute__((ext_vector_type(8)))  float         v8f;
typedef __attribute__((ext_vector_type(4)))  float         f32x4;
typedef __attribute__((ext_vector_type(4)))  int           v4i;
typedef __attribute__((ext_vector_type(4)))  unsigned int  u32x4;
typedef __attribute__((ext_vector_type(2)))  unsigned int  u32x2;

#if __has_builtin(__builtin_amdgcn_global_load_async_to_lds_b128) && \
    __has_builtin(__builtin_amdgcn_s_wait_asynccnt)
#define HAS_ASYNC 1
#define AS1 __attribute__((address_space(1)))
#define AS3 __attribute__((address_space(3)))
#endif

union BF16Frag {
  v16bf          v;
  u32x4          q[2];
  unsigned short s[16];
};

__device__ __forceinline__ unsigned short f2bf(float f) {
  unsigned int u = __float_as_uint(f);
  u += 0x7FFFu + ((u >> 16) & 1u);          // round-to-nearest-even
  return (unsigned short)(u >> 16);
}

#if __has_builtin(__builtin_amdgcn_cvt_pk_bf16_f32)
__device__ __forceinline__ unsigned int pk2(float a, float b) {
  union { v2bf v; unsigned int u; } cv;
  cv.v = __builtin_amdgcn_cvt_pk_bf16_f32(a, b);
  return cv.u;
}
#else
__device__ __forceinline__ unsigned int pk2(float a, float b) {
  return (unsigned int)f2bf(a) | ((unsigned int)f2bf(b) << 16);
}
#endif

// ---------------------------------------------------------------------------
// Kernel 0: transpose + convert weights to bf16, [N][K] layout (K-contiguous)
// ---------------------------------------------------------------------------
__global__ __launch_bounds__(256) void prep_weights(
    const float* __restrict__ Wqkv, const float* __restrict__ Wproj,
    unsigned short* __restrict__ WqkvT, unsigned short* __restrict__ WprojT) {
  const int stride = gridDim.x * blockDim.x;
  int i = blockIdx.x * blockDim.x + threadIdx.x;
  const int total1 = 3 * EMBED * EMBED;
  for (int o = i; o < total1; o += stride) {
    int n = o >> 10, k = o & 1023;
    WqkvT[o] = f2bf(Wqkv[(size_t)k * (3 * EMBED) + n]);
  }
  const int total2 = EMBED * EMBED;
  for (int o = i; o < total2; o += stride) {
    int n = o >> 10, k = o & 1023;
    WprojT[o] = f2bf(Wproj[(size_t)k * EMBED + n]);
  }
}

// ---------------------------------------------------------------------------
// Kernel 1: qkv = x @ W_qkv + b, scattered to bf16 Q[B,H,S,D], K[B,H,S,D],
//           and transposed V^T[B,H,D,S].
// ---------------------------------------------------------------------------
__global__ __launch_bounds__(256) void qkv_gemm(
    const float* __restrict__ x, const unsigned short* __restrict__ Wt,
    const float* __restrict__ bias, unsigned short* __restrict__ qb,
    unsigned short* __restrict__ kb, unsigned short* __restrict__ vt) {
  __shared__ __attribute__((aligned(16))) unsigned short sA[128 * 32];
  __shared__ __attribute__((aligned(16))) unsigned short sB[128 * 32];
  const int tid = threadIdx.x;
  const int lane = tid & 31, wid = tid >> 5;
  const int h = lane >> 4, lr = lane & 15;
  const int m0 = blockIdx.y * 128, n0 = blockIdx.x * 128;
  const int wm0 = (wid & 3) * 32, wn0 = (wid >> 2) * 64;

  v8f acc[2][4];
#pragma unroll
  for (int mi = 0; mi < 2; ++mi)
#pragma unroll
    for (int ni = 0; ni < 4; ++ni)
#pragma unroll
      for (int r = 0; r < 8; ++r) acc[mi][ni][r] = 0.f;

  for (int kk = 0; kk < EMBED; kk += 32) {
    // --- B tile: bf16, direct copy (async LDS path when available) ---
#ifdef HAS_ASYNC
#pragma unroll
    for (int it = 0; it < 2; ++it) {
      int idx = tid + it * 256;                    // uint4 index [0,512)
      int r = idx >> 2, c = (idx & 3) << 3;
      __builtin_amdgcn_global_load_async_to_lds_b128(
          (AS1 v4i*)(Wt + (size_t)(n0 + r) * EMBED + kk + c),
          (AS3 v4i*)&sB[r * 32 + c], 0, 0);
    }
#else
    u32x4 vb[2];
#pragma unroll
    for (int it = 0; it < 2; ++it) {
      int idx = tid + it * 256;
      int r = idx >> 2, c = (idx & 3) << 3;
      vb[it] = *(const u32x4*)(Wt + (size_t)(n0 + r) * EMBED + kk + c);
    }
#endif
    // --- A tile: fp32 -> bf16: issue all loads, then convert/store ---
    f32x4 va[4];
#pragma unroll
    for (int it = 0; it < 4; ++it) {
      int idx = tid + it * 256;                    // float4 index [0,1024)
      int r = idx >> 3, c = (idx & 7) << 2;
      va[it] = *(const f32x4*)(x + (size_t)(m0 + r) * EMBED + kk + c);
    }
#pragma unroll
    for (int it = 0; it < 4; ++it) {
      int idx = tid + it * 256;
      int r = idx >> 3, c = (idx & 7) << 2;
      u32x2 p; p.x = pk2(va[it].x, va[it].y); p.y = pk2(va[it].z, va[it].w);
      *(u32x2*)(&sA[r * 32 + c]) = p;
    }
#ifndef HAS_ASYNC
#pragma unroll
    for (int it = 0; it < 2; ++it) {
      int idx = tid + it * 256;
      int r = idx >> 2, c = (idx & 3) << 3;
      *(u32x4*)(&sB[r * 32 + c]) = vb[it];
    }
#endif
#ifdef HAS_ASYNC
    __builtin_amdgcn_s_wait_asynccnt(0);
#endif
    __syncthreads();

    BF16Frag a[2], b[4];
#pragma unroll
    for (int mi = 0; mi < 2; ++mi) {
      int r = wm0 + mi * 16 + lr;
      a[mi].q[0] = *(const u32x4*)(&sA[r * 32 + h * 8]);
      a[mi].q[1] = *(const u32x4*)(&sA[r * 32 + 16 + h * 8]);
    }
#pragma unroll
    for (int ni = 0; ni < 4; ++ni) {
      int r = wn0 + ni * 16 + lr;
      b[ni].q[0] = *(const u32x4*)(&sB[r * 32 + h * 8]);
      b[ni].q[1] = *(const u32x4*)(&sB[r * 32 + 16 + h * 8]);
    }
#pragma unroll
    for (int mi = 0; mi < 2; ++mi)
#pragma unroll
      for (int ni = 0; ni < 4; ++ni)
        acc[mi][ni] = __builtin_amdgcn_wmma_f32_16x16x32_bf16(
            false, a[mi].v, false, b[ni].v, (short)0, acc[mi][ni], false, false);
    __syncthreads();
  }

  // Epilogue: bias, convert, scatter into Q / K / V^T
#pragma unroll
  for (int mi = 0; mi < 2; ++mi)
#pragma unroll
    for (int ni = 0; ni < 4; ++ni) {
      int gn = n0 + wn0 + ni * 16 + lr;
      float bv = bias[gn];
      int part = gn >> 10, e = gn & 1023, head = e >> 6, dd = e & 63;
#pragma unroll
      for (int r = 0; r < 8; ++r) {
        int gm = m0 + wm0 + mi * 16 + r + 8 * h;
        int bq = gm >> 11, sq = gm & 2047;
        int bhh = bq * NHEAD + head;
        unsigned short o = f2bf(acc[mi][ni][r] + bv);
        if (part == 0)
          qb[((size_t)(bhh * SEQ + sq) << 6) + dd] = o;
        else if (part == 1)
          kb[((size_t)(bhh * SEQ + sq) << 6) + dd] = o;
        else
          vt[((size_t)(bhh * HDIM + dd) << 11) + sq] = o;
      }
    }
}

// ---------------------------------------------------------------------------
// Kernel 2: fused attention. One WG per (b*h, 32-query block).
// ---------------------------------------------------------------------------
__global__ __launch_bounds__(256) void attn_fused(
    const unsigned short* __restrict__ qb, const unsigned short* __restrict__ kbuf,
    const unsigned short* __restrict__ vt, float* __restrict__ attn,
    unsigned short* __restrict__ ctxb) {
  __shared__ __attribute__((aligned(16))) unsigned short sQ[32 * 64];
  __shared__ __attribute__((aligned(16))) float sS[32 * SST];
  const int tid = threadIdx.x, lane = tid & 31, wid = tid >> 5;
  const int h = lane >> 4, lr = lane & 15;
  const int bh = blockIdx.x >> 6;            // [0, B*H)
  const int q0 = (blockIdx.x & 63) * 32;
  const size_t qkbase = (size_t)bh * SEQ * HDIM;
  const size_t vbase  = (size_t)bh * HDIM * SEQ;

  {  // load Q tile 32x64 bf16
    int r = tid >> 3, c = (tid & 7) << 3;
    *(u32x4*)(&sQ[r * 64 + c]) =
        *(const u32x4*)(qb + qkbase + (size_t)(q0 + r) * HDIM + c);
  }
  __syncthreads();

  // ---- Phase 1: scores -> LDS ----
  BF16Frag aq[2][2];
#pragma unroll
  for (int mi = 0; mi < 2; ++mi)
#pragma unroll
    for (int kbi = 0; kbi < 2; ++kbi) {
      int r = mi * 16 + lr;
      aq[mi][kbi].q[0] = *(const u32x4*)(&sQ[r * 64 + kbi * 32 + h * 8]);
      aq[mi][kbi].q[1] = *(const u32x4*)(&sQ[r * 64 + kbi * 32 + 16 + h * 8]);
    }
  const int nbase = wid * 256;               // per-wave key stripe

  auto loadK = [&](int t, BF16Frag* dst) {
    int scol = nbase + t * 16;
#pragma unroll
    for (int kbi = 0; kbi < 2; ++kbi) {
      const unsigned short* kp =
          kbuf + qkbase + (size_t)(scol + lr) * HDIM + kbi * 32 + h * 8;
      dst[kbi].q[0] = *(const u32x4*)(kp);
      dst[kbi].q[1] = *(const u32x4*)(kp + 16);
    }
  };

  BF16Frag bk[2];
  loadK(0, bk);
  for (int t = 0; t < 16; ++t) {
    BF16Frag nbk[2];
    if (t < 15) loadK(t + 1, nbk);           // prefetch next tile over WMMA
    int scol = nbase + t * 16;
    v8f a0, a1;
#pragma unroll
    for (int r = 0; r < 8; ++r) { a0[r] = 0.f; a1[r] = 0.f; }
#pragma unroll
    for (int kbi = 0; kbi < 2; ++kbi) {
      a0 = __builtin_amdgcn_wmma_f32_16x16x32_bf16(
          false, aq[0][kbi].v, false, bk[kbi].v, (short)0, a0, false, false);
      a1 = __builtin_amdgcn_wmma_f32_16x16x32_bf16(
          false, aq[1][kbi].v, false, bk[kbi].v, (short)0, a1, false, false);
    }
#pragma unroll
    for (int r = 0; r < 8; ++r) {
      sS[(r + 8 * h) * SST + scol + lr]      = a0[r] * 0.125f;
      sS[(16 + r + 8 * h) * SST + scol + lr] = a1[r] * 0.125f;
    }
    bk[0] = nbk[0]; bk[1] = nbk[1];
  }
  __syncthreads();

  // ---- Phase 2: softmax (4 rows per wave), write attn fp32 ----
  for (int rr = 0; rr < 4; ++rr) {
    int row = wid * 4 + rr;
    float* rp = &sS[row * SST];
    float m = -3.402823466e38f;
#pragma unroll
    for (int i = 0; i < 64; ++i) m = fmaxf(m, rp[lane + 32 * i]);
#pragma unroll
    for (int off = 16; off > 0; off >>= 1) m = fmaxf(m, __shfl_xor(m, off, 32));
    float sum = 0.f;
#pragma unroll
    for (int i = 0; i < 64; ++i) {
      float e = __expf(rp[lane + 32 * i] - m);
      rp[lane + 32 * i] = e;
      sum += e;
    }
#pragma unroll
    for (int off = 16; off > 0; off >>= 1) sum += __shfl_xor(sum, off, 32);
    float inv = 1.0f / sum;
    float* arow = attn + ((size_t)(bh * SEQ + q0 + row)) * SEQ;
#pragma unroll
    for (int i = 0; i < 64; ++i) {
      float p = rp[lane + 32 * i] * inv;
      rp[lane + 32 * i] = p;
      arow[lane + 32 * i] = p;
    }
  }
  __syncthreads();

  // ---- Phase 3: ctx = P @ V (per-wave K stripe of 256) ----
  v8f acc[2][4];
#pragma unroll
  for (int mi = 0; mi < 2; ++mi)
#pragma unroll
    for (int ni = 0; ni < 4; ++ni)
#pragma unroll
      for (int r = 0; r < 8; ++r) acc[mi][ni][r] = 0.f;

  const int sbase = wid * 256;
  auto loadV = [&](int kt, BF16Frag* dst) {
    int ks = sbase + kt * 32;
#pragma unroll
    for (int ni = 0; ni < 4; ++ni) {
      const unsigned short* vp =
          vt + vbase + (size_t)(ni * 16 + lr) * SEQ + ks + h * 8;
      dst[ni].q[0] = *(const u32x4*)(vp);
      dst[ni].q[1] = *(const u32x4*)(vp + 16);
    }
  };

  BF16Frag bv[4];
  loadV(0, bv);
  for (int kt = 0; kt < 8; ++kt) {
    BF16Frag nbv[4];
    if (kt < 7) loadV(kt + 1, nbv);          // prefetch next V tile over WMMA
    int ks = sbase + kt * 32;
    BF16Frag ap[2];
#pragma unroll
    for (int mi = 0; mi < 2; ++mi) {
      const float* p0 = &sS[(mi * 16 + lr) * SST + ks + h * 8];
      const float* p1 = &sS[(mi * 16 + lr) * SST + ks + 16 + h * 8];
      f32x4 f0 = *(const f32x4*)(p0);
      f32x4 f1 = *(const f32x4*)(p0 + 4);
      f32x4 f2 = *(const f32x4*)(p1);
      f32x4 f3 = *(const f32x4*)(p1 + 4);
      u32x4 lo, hi;
      lo.x = pk2(f0.x, f0.y); lo.y = pk2(f0.z, f0.w);
      lo.z = pk2(f1.x, f1.y); lo.w = pk2(f1.z, f1.w);
      hi.x = pk2(f2.x, f2.y); hi.y = pk2(f2.z, f2.w);
      hi.z = pk2(f3.x, f3.y); hi.w = pk2(f3.z, f3.w);
      ap[mi].q[0] = lo; ap[mi].q[1] = hi;
    }
#pragma unroll
    for (int mi = 0; mi < 2; ++mi)
#pragma unroll
      for (int ni = 0; ni < 4; ++ni)
        acc[mi][ni] = __builtin_amdgcn_wmma_f32_16x16x32_bf16(
            false, ap[mi].v, false, bv[ni].v, (short)0, acc[mi][ni], false, false);
#pragma unroll
    for (int ni = 0; ni < 4; ++ni) bv[ni] = nbv[ni];
  }

  // Deterministic cross-wave reduction: reuse score LDS as 8 x 2048 floats
  __syncthreads();                 // all P reads complete before overwrite
  float* pbuf = sS;
#pragma unroll
  for (int mi = 0; mi < 2; ++mi)
#pragma unroll
    for (int ni = 0; ni < 4; ++ni)
#pragma unroll
      for (int r = 0; r < 8; ++r)
        pbuf[wid * 2048 + (mi * 16 + r + 8 * h) * 64 + ni * 16 + lr] =
            acc[mi][ni][r];
  __syncthreads();

  const int b = bh >> 4, hh = bh & 15;
#pragma unroll
  for (int it = 0; it < 8; ++it) {
    int idx = tid + it * 256;      // [0, 2048): row*64 + d
    float s = 0.f;
#pragma unroll
    for (int w = 0; w < 8; ++w) s += pbuf[w * 2048 + idx];
    int r = idx >> 6, d = idx & 63;
    ctxb[(size_t)(b * SEQ + q0 + r) * EMBED + hh * HDIM + d] = f2bf(s);
  }
}

// ---------------------------------------------------------------------------
// Kernel 3: out = ctx @ W_proj + b_proj (fp32 out).
// ---------------------------------------------------------------------------
__global__ __launch_bounds__(256) void proj_gemm(
    const unsigned short* __restrict__ A, const unsigned short* __restrict__ Wt,
    const float* __restrict__ bias, float* __restrict__ out) {
  __shared__ __attribute__((aligned(16))) unsigned short sA[128 * 32];
  __shared__ __attribute__((aligned(16))) unsigned short sB[128 * 32];
  const int tid = threadIdx.x;
  const int lane = tid & 31, wid = tid >> 5;
  const int h = lane >> 4, lr = lane & 15;
  const int m0 = blockIdx.y * 128, n0 = blockIdx.x * 128;
  const int wm0 = (wid & 3) * 32, wn0 = (wid >> 2) * 64;

  v8f acc[2][4];
#pragma unroll
  for (int mi = 0; mi < 2; ++mi)
#pragma unroll
    for (int ni = 0; ni < 4; ++ni)
#pragma unroll
      for (int r = 0; r < 8; ++r) acc[mi][ni][r] = 0.f;

  for (int kk = 0; kk < EMBED; kk += 32) {
#ifdef HAS_ASYNC
#pragma unroll
    for (int it = 0; it < 2; ++it) {
      int idx = tid + it * 256;
      int r = idx >> 2, c = (idx & 3) << 3;
      __builtin_amdgcn_global_load_async_to_lds_b128(
          (AS1 v4i*)(A + (size_t)(m0 + r) * EMBED + kk + c),
          (AS3 v4i*)&sA[r * 32 + c], 0, 0);
      __builtin_amdgcn_global_load_async_to_lds_b128(
          (AS1 v4i*)(Wt + (size_t)(n0 + r) * EMBED + kk + c),
          (AS3 v4i*)&sB[r * 32 + c], 0, 0);
    }
    __builtin_amdgcn_s_wait_asynccnt(0);
#else
    u32x4 va[2], vb[2];
#pragma unroll
    for (int it = 0; it < 2; ++it) {
      int idx = tid + it * 256;
      int r = idx >> 2, c = (idx & 3) << 3;
      va[it] = *(const u32x4*)(A + (size_t)(m0 + r) * EMBED + kk + c);
      vb[it] = *(const u32x4*)(Wt + (size_t)(n0 + r) * EMBED + kk + c);
    }
#pragma unroll
    for (int it = 0; it < 2; ++it) {
      int idx = tid + it * 256;
      int r = idx >> 2, c = (idx & 3) << 3;
      *(u32x4*)(&sA[r * 32 + c]) = va[it];
      *(u32x4*)(&sB[r * 32 + c]) = vb[it];
    }
#endif
    __syncthreads();

    BF16Frag a[2], b[4];
#pragma unroll
    for (int mi = 0; mi < 2; ++mi) {
      int r = wm0 + mi * 16 + lr;
      a[mi].q[0] = *(const u32x4*)(&sA[r * 32 + h * 8]);
      a[mi].q[1] = *(const u32x4*)(&sA[r * 32 + 16 + h * 8]);
    }
#pragma unroll
    for (int ni = 0; ni < 4; ++ni) {
      int r = wn0 + ni * 16 + lr;
      b[ni].q[0] = *(const u32x4*)(&sB[r * 32 + h * 8]);
      b[ni].q[1] = *(const u32x4*)(&sB[r * 32 + 16 + h * 8]);
    }
#pragma unroll
    for (int mi = 0; mi < 2; ++mi)
#pragma unroll
      for (int ni = 0; ni < 4; ++ni)
        acc[mi][ni] = __builtin_amdgcn_wmma_f32_16x16x32_bf16(
            false, a[mi].v, false, b[ni].v, (short)0, acc[mi][ni], false, false);
    __syncthreads();
  }

#pragma unroll
  for (int mi = 0; mi < 2; ++mi)
#pragma unroll
    for (int ni = 0; ni < 4; ++ni) {
      int gn = n0 + wn0 + ni * 16 + lr;
      float bv = bias[gn];
#pragma unroll
      for (int r = 0; r < 8; ++r) {
        int gm = m0 + wm0 + mi * 16 + r + 8 * h;
        out[(size_t)gm * EMBED + gn] = acc[mi][ni][r] + bv;
      }
    }
}

// ---------------------------------------------------------------------------
extern "C" void kernel_launch(void* const* d_in, const int* in_sizes, int n_in,
                              void* d_out, int out_size, void* d_ws, size_t ws_size,
                              hipStream_t stream) {
  (void)in_sizes; (void)n_in; (void)out_size; (void)ws_size;
  const float* x     = (const float*)d_in[0];
  const float* Wqkv  = (const float*)d_in[1];
  const float* bqkv  = (const float*)d_in[2];
  const float* Wproj = (const float*)d_in[3];
  const float* bproj = (const float*)d_in[4];

  float* out  = (float*)d_out;                       // [B,S,E] fp32
  float* attn = out + (size_t)NTOK * EMBED;          // [B,H,S,S] fp32

  char* ws = (char*)d_ws;
  unsigned short* WqkvT = (unsigned short*)(ws);                 // 6 MB  [3072][1024] bf16
  unsigned short* WprojT= (unsigned short*)(ws + 6291456);       // 2 MB  [1024][1024] bf16
  unsigned short* qbf   = (unsigned short*)(ws + 8388608);       // 16 MB [B,H,S,D] bf16
  unsigned short* kbf   = (unsigned short*)(ws + 25165824);      // 16 MB [B,H,S,D] bf16
  unsigned short* vtb   = (unsigned short*)(ws + 41943040);      // 16 MB [B,H,D,S] bf16
  unsigned short* ctxb  = (unsigned short*)(ws + 58720256);      // 16 MB [B*S][E]   bf16

  prep_weights<<<dim3(2048), dim3(256), 0, stream>>>(Wqkv, Wproj, WqkvT, WprojT);
  qkv_gemm   <<<dim3(24, 64), dim3(256), 0, stream>>>(x, WqkvT, bqkv, qbf, kbf, vtb);
  attn_fused <<<dim3(4096),   dim3(256), 0, stream>>>(qbf, kbf, vtb, attn, ctxb);
  proj_gemm  <<<dim3(8, 64),  dim3(256), 0, stream>>>(ctxb, WprojT, bproj, out);
}
